// CandidateGenerator_20100446945309
// MI455X (gfx1250) — compile-verified, matrix operand order checked by reading, not run on hardware
//
#include <hip/hip_runtime.h>
#include <stdint.h>

// ---------------------------------------------------------------------------
// CDNA5 (gfx1250) candidate generator:
//   K1 qproj:  hidden@Wq (bf16 WMMA) -> LayerNorm -> tanh -> qv, qv_bf16, qcodes
//   K2 hash:   item_emb -> bf16 copy + packed 4x8-bit LSH codes (bf16 WMMA + ballot)
//   K3 score:  async-staged item tiles (global_load_async_to_lds_b128) ->
//              qv_bf @ items_bf^T (bf16 WMMA), packed-code byte match,
//              append matched candidates + 1 filler/block to per-query buffers
//   K4 select: exact top-100 per query (iterative max over <=6144 entries in LDS)
//   K5 gather: candidate embeddings
// ---------------------------------------------------------------------------

typedef __attribute__((ext_vector_type(16))) __bf16 v16bf;
typedef __attribute__((ext_vector_type(8)))  float  v8f;

union Frag {
  v16bf v;
  uint4 q[2];
  unsigned short u[16];
};

static constexpr int CAPC = 6144;   // per-query candidate buffer capacity
static constexpr int KSEL = 100;

__device__ __forceinline__ unsigned short f2bf(float f) {
  unsigned int u = __builtin_bit_cast(unsigned int, f);
  unsigned int r = (u + 0x7FFFu + ((u >> 16) & 1u)) >> 16;   // RNE
  return (unsigned short)r;
}
__device__ __forceinline__ unsigned int fkey(float f) {       // order-preserving uint
  unsigned int u = __builtin_bit_cast(unsigned int, f);
  return (u & 0x80000000u) ? ~u : (u | 0x80000000u);
}
__device__ __forceinline__ float keyf(unsigned int k) {
  unsigned int u = (k & 0x80000000u) ? (k ^ 0x80000000u) : ~k;
  return __builtin_bit_cast(float, u);
}

// -------------------------------- K1 ---------------------------------------
__global__ __launch_bounds__(256) void qproj_kernel(
    const float* __restrict__ hidden, const float* __restrict__ Wq,
    const float* __restrict__ bq, const float* __restrict__ gamma,
    const float* __restrict__ beta, const float* __restrict__ proj,
    float* __restrict__ qv_out, unsigned short* __restrict__ qv_bf,
    unsigned int* __restrict__ qcodes)
{
  __shared__ float qt[16][256];
  const int tid = threadIdx.x, wave = tid >> 5, lane = tid & 31;
  const int half = lane >> 4, l15 = lane & 15;
  const int qbase = blockIdx.x * 16;
  const int n0 = wave * 32;                      // 8 waves * 32 cols = 256 cols

  v8f c0 = {}; v8f c1 = {};
  for (int k = 0; k < 2048; k += 32) {
    Frag a;
    {
      const float* hp = hidden + (size_t)(qbase + l15) * 2048 + k + half * 8;
      float4 f0 = *(const float4*)(hp);
      float4 f1 = *(const float4*)(hp + 4);
      float4 f2 = *(const float4*)(hp + 16);
      float4 f3 = *(const float4*)(hp + 20);
      a.u[0]=f2bf(f0.x); a.u[1]=f2bf(f0.y); a.u[2]=f2bf(f0.z); a.u[3]=f2bf(f0.w);
      a.u[4]=f2bf(f1.x); a.u[5]=f2bf(f1.y); a.u[6]=f2bf(f1.z); a.u[7]=f2bf(f1.w);
      a.u[8]=f2bf(f2.x); a.u[9]=f2bf(f2.y); a.u[10]=f2bf(f2.z); a.u[11]=f2bf(f2.w);
      a.u[12]=f2bf(f3.x); a.u[13]=f2bf(f3.y); a.u[14]=f2bf(f3.z); a.u[15]=f2bf(f3.w);
    }
    Frag b0, b1;
    const int kb = k + half * 8;
#pragma unroll
    for (int j = 0; j < 8; ++j) {
      b0.u[j]     = f2bf(Wq[(size_t)(kb + j)      * 256 + n0 + l15]);
      b0.u[8 + j] = f2bf(Wq[(size_t)(kb + 16 + j) * 256 + n0 + l15]);
      b1.u[j]     = f2bf(Wq[(size_t)(kb + j)      * 256 + n0 + 16 + l15]);
      b1.u[8 + j] = f2bf(Wq[(size_t)(kb + 16 + j) * 256 + n0 + 16 + l15]);
    }
    c0 = __builtin_amdgcn_wmma_f32_16x16x32_bf16(false, a.v, false, b0.v, (short)0, c0, false, false);
    c1 = __builtin_amdgcn_wmma_f32_16x16x32_bf16(false, a.v, false, b1.v, (short)0, c1, false, false);
  }
#pragma unroll
  for (int r = 0; r < 8; ++r) {
    const int m = r + half * 8;
    qt[m][n0 + l15]      = c0[r] + bq[n0 + l15];
    qt[m][n0 + 16 + l15] = c1[r] + bq[n0 + 16 + l15];
  }
  __syncthreads();

  // LayerNorm + tanh: wave handles rows {wave, wave+8}
  for (int rr = 0; rr < 2; ++rr) {
    const int row = wave + rr * 8;
    float s = 0.f;
#pragma unroll
    for (int j = 0; j < 8; ++j) s += qt[row][lane + 32 * j];
#pragma unroll
    for (int o = 16; o > 0; o >>= 1) s += __shfl_xor(s, o, 32);
    const float mu = s * (1.0f / 256.0f);
    float var = 0.f;
#pragma unroll
    for (int j = 0; j < 8; ++j) { float d = qt[row][lane + 32 * j] - mu; var += d * d; }
#pragma unroll
    for (int o = 16; o > 0; o >>= 1) var += __shfl_xor(var, o, 32);
    const float rs = rsqrtf(var * (1.0f / 256.0f) + 1e-5f);
#pragma unroll
    for (int j = 0; j < 8; ++j) {
      const int col = lane + 32 * j;
      const float y = (qt[row][col] - mu) * rs * gamma[col] + beta[col];
      const float t = tanhf(y);
      qt[row][col] = t;
      qv_out[(size_t)(qbase + row) * 256 + col] = t;
      qv_bf[(size_t)(qbase + row) * 256 + col] = f2bf(t);
    }
  }
  __syncthreads();

  // Query LSH codes: lane = t*8 + bit; ballot mask IS the packed 32-bit code.
  for (int rr = 0; rr < 2; ++rr) {
    const int lrow = wave + rr * 8;
    const int t = lane >> 3, bit = lane & 7;
    float d = 0.f;
    for (int e = 0; e < 256; ++e) d += qt[lrow][e] * proj[t * 2048 + e * 8 + bit];
    unsigned long long m = __ballot(d > 0.0f);
    if (lane == 0) qcodes[qbase + lrow] = (unsigned int)m;
  }
}

// -------------------------------- K2 ---------------------------------------
__global__ __launch_bounds__(256) void item_hash_kernel(
    const float* __restrict__ item_emb, const float* __restrict__ proj,
    unsigned short* __restrict__ itemsBF, unsigned int* __restrict__ itemCodes,
    int N)
{
  __shared__ unsigned short projB[256 * 32];     // B layout [e][col], 16 KB
  __shared__ __align__(16) unsigned short it[64 * 256];  // 32 KB item tile, bf16
  __shared__ unsigned short pc16[64][2];
  const int tid = threadIdx.x, wave = tid >> 5, lane = tid & 31;
  const int half = lane >> 4, l15 = lane & 15;
  const int ib = blockIdx.x * 64;

  for (int i = tid; i < 256 * 32; i += 256) {
    const int e = i >> 5, col = i & 31;          // col = t*8 + bit
    projB[i] = f2bf(proj[(col >> 3) * 2048 + e * 8 + (col & 7)]);
  }
  for (int i = tid; i < 64 * 256; i += 256) {
    const int row = i >> 8, col = i & 255;
    int g = ib + row; if (g >= N) g = N - 1;
    const unsigned short h = f2bf(item_emb[(size_t)g * 256 + col]);
    it[i] = h;
    itemsBF[(size_t)(ib + row) * 256 + col] = h; // bf16 item copy (L2-resident)
  }
  __syncthreads();

  const int mt = wave >> 1, nt = wave & 1;       // 4 m-tiles x 2 n-tiles
  v8f c = {};
#pragma unroll
  for (int ks = 0; ks < 8; ++ks) {
    const int kb = ks * 32 + half * 8;
    Frag a, b;
    const unsigned short* ap = &it[(mt * 16 + l15) * 256 + kb];
    a.q[0] = *(const uint4*)(ap);
    a.q[1] = *(const uint4*)(ap + 16);
#pragma unroll
    for (int j = 0; j < 8; ++j) {
      b.u[j]     = projB[(kb + j)      * 32 + nt * 16 + l15];
      b.u[8 + j] = projB[(kb + 16 + j) * 32 + nt * 16 + l15];
    }
    c = __builtin_amdgcn_wmma_f32_16x16x32_bf16(false, a.v, false, b.v, (short)0, c, false, false);
  }
  unsigned int m[8];
#pragma unroll
  for (int r = 0; r < 8; ++r) m[r] = (unsigned int)__ballot(c[r] > 0.0f);
  if (lane == 0) {
#pragma unroll
    for (int r = 0; r < 8; ++r) {
      pc16[mt * 16 + r][nt]     = (unsigned short)(m[r] & 0xFFFFu); // row r
      pc16[mt * 16 + r + 8][nt] = (unsigned short)(m[r] >> 16);     // row r+8
    }
  }
  __syncthreads();
  if (tid < 64) {
    const int g = ib + tid;
    if (g < N)
      itemCodes[g] = (unsigned int)pc16[tid][0] | ((unsigned int)pc16[tid][1] << 16);
  }
}

// -------------------------------- K3 ---------------------------------------
// Stage 64 items x 256 bf16 (32 KB) per step via async global->LDS DMA; 8 waves
// split as 4 n-tiles x 2 K-halves; kh=1 partial C combined through LDS.
__global__ __launch_bounds__(256) void score_kernel(
    const unsigned short* __restrict__ qv_bf, const unsigned int* __restrict__ qcodes,
    const unsigned short* __restrict__ itemsBF, const unsigned int* __restrict__ itemCodes,
    float* __restrict__ candScore, int* __restrict__ candId, int* __restrict__ candCnt,
    int N)
{
  __shared__ __align__(16) unsigned short tile[64 * 256];  // 32 KB bf16 items
  __shared__ float pcs[4][16][16];                         // kh=1 partial C, 4 KB
  __shared__ unsigned long long blockBest[16];
  const int tid = threadIdx.x, wave = tid >> 5, lane = tid & 31;
  const int half = lane >> 4, l15 = lane & 15;
  const int nt = wave & 3, kh = wave >> 2;
  const int qb = blockIdx.y;                               // 16-query block
  const long long chunk = (long long)blockIdx.x * 2048;
  if (tid < 16) blockBest[tid] = 0ull;

  Frag af[4];                                              // this wave's K-half
#pragma unroll
  for (int ks = 0; ks < 4; ++ks) {
    const int kglob = kh * 128 + ks * 32;
    const unsigned short* p = qv_bf + (size_t)(qb * 16 + l15) * 256 + kglob + half * 8;
    af[ks].q[0] = *(const uint4*)(p);
    af[ks].q[1] = *(const uint4*)(p + 16);
  }
  unsigned int qcd[8];
#pragma unroll
  for (int r = 0; r < 8; ++r) qcd[r] = qcodes[qb * 16 + r + 8 * half];

  unsigned long long best[8];
#pragma unroll
  for (int r = 0; r < 8; ++r) best[r] = 0ull;

  const unsigned myLds = (unsigned)(uintptr_t)(&tile[0]) + (unsigned)(tid * 16);

  for (int itr = 0; itr < 32; ++itr) {                     // 32 x 64 = 2048 items
    const long long nstage = chunk + (long long)itr * 64;  // block-uniform
    if (nstage >= N) break;

    // --- async DMA: 256 threads x 8 x 16B = 32 KB tile ---
    {
      const unsigned long long ga =
          (unsigned long long)(uintptr_t)(itemsBF + nstage * 256) +
          (unsigned long long)(tid * 16);
      asm volatile(
          "global_load_async_to_lds_b128 %0, %1, off\n\t"
          "global_load_async_to_lds_b128 %0, %1, off offset:4096\n\t"
          "global_load_async_to_lds_b128 %0, %1, off offset:8192\n\t"
          "global_load_async_to_lds_b128 %0, %1, off offset:12288\n\t"
          "global_load_async_to_lds_b128 %0, %1, off offset:16384\n\t"
          "global_load_async_to_lds_b128 %0, %1, off offset:20480\n\t"
          "global_load_async_to_lds_b128 %0, %1, off offset:24576\n\t"
          "global_load_async_to_lds_b128 %0, %1, off offset:28672\n\t"
          :: "v"(myLds), "v"(ga) : "memory");
      asm volatile("s_wait_asynccnt 0x0" ::: "memory");
    }
    __syncthreads();                                       // tile visible to WGP

    v8f c = {};
#pragma unroll
    for (int ks = 0; ks < 4; ++ks) {
      Frag b;
      const unsigned short* p = &tile[(nt * 16 + l15) * 256 + kh * 128 + ks * 32 + half * 8];
      b.q[0] = *(const uint4*)(p);
      b.q[1] = *(const uint4*)(p + 16);
      c = __builtin_amdgcn_wmma_f32_16x16x32_bf16(false, af[ks].v, false, b.v, (short)0, c, false, false);
    }
    if (kh == 1) {
#pragma unroll
      for (int r = 0; r < 8; ++r) pcs[nt][r + 8 * half][l15] = c[r];
    }
    __syncthreads();                                       // partials ready

    if (kh == 0) {
      const int n = (int)nstage + nt * 16 + l15;
      const int nc = n < N ? n : N - 1;
      const unsigned int icode = itemCodes[nc];
      const bool valid = (n < N);
#pragma unroll
      for (int r = 0; r < 8; ++r) {
        const float s = c[r] + pcs[nt][r + 8 * half][l15];
        const unsigned int x = qcd[r] ^ icode;
        const bool matched = ((x - 0x01010101u) & ~x & 0x80808080u) != 0u;
        if (valid) {
          if (matched) {
            const int qrow = qb * 16 + r + 8 * half;
            const int idx = atomicAdd(&candCnt[qrow], 1);
            if (idx < CAPC) {
              candScore[(size_t)qrow * CAPC + idx] = s;
              candId[(size_t)qrow * CAPC + idx]   = n;
            }
          } else {
            const unsigned long long key =
                ((unsigned long long)fkey(s - 1.0e6f) << 32) | (unsigned int)n;
            if (key > best[r]) best[r] = key;
          }
        }
      }
    }
    __syncthreads();                                       // tile/pcs reusable
  }

  // best unmatched filler per query row (deterministic pad when candidates < K)
#pragma unroll
  for (int r = 0; r < 8; ++r) {
    unsigned long long v = best[r];
#pragma unroll
    for (int o = 8; o > 0; o >>= 1) {
      unsigned long long w = (unsigned long long)__shfl_xor((long long)v, o, 16);
      if (w > v) v = w;
    }
    if (l15 == 0 && v) atomicMax(&blockBest[r + 8 * half], v);
  }
  __syncthreads();
  if (tid < 16) {
    const unsigned long long v = blockBest[tid];
    if (v) {
      const int qrow = qb * 16 + tid;
      const int idx = atomicAdd(&candCnt[qrow], 1);
      if (idx < CAPC) {
        candScore[(size_t)qrow * CAPC + idx] = keyf((unsigned int)(v >> 32));
        candId[(size_t)qrow * CAPC + idx]   = (int)(unsigned int)v;
      }
    }
  }
}

// -------------------------------- K4 ---------------------------------------
__global__ __launch_bounds__(256) void select_kernel(
    const float* __restrict__ candScore, const int* __restrict__ candId,
    const int* __restrict__ candCnt, float* __restrict__ ids_out,
    int* __restrict__ selIds)
{
  __shared__ unsigned long long keys[CAPC];   // 48 KB
  __shared__ unsigned long long sKey[256];
  __shared__ int sIdx[256];
  const int q = blockIdx.x, tid = threadIdx.x;
  int cnt = candCnt[q]; if (cnt > CAPC) cnt = CAPC;
  for (int i = tid; i < CAPC; i += 256) {
    if (i < cnt) {
      keys[i] = ((unsigned long long)fkey(candScore[(size_t)q * CAPC + i]) << 32)
              | (unsigned int)(~(unsigned int)candId[(size_t)q * CAPC + i]); // tie: lower id wins
    } else {
      keys[i] = ((unsigned long long)fkey(-3.0e38f) << 32) | 0xFFFFFFFFu;    // pad -> id 0
    }
  }
  __syncthreads();
  for (int k = 0; k < KSEL; ++k) {
    unsigned long long bk = 0ull; int bi = 0;
    for (int i = tid; i < CAPC; i += 256)
      if (keys[i] > bk) { bk = keys[i]; bi = i; }
    sKey[tid] = bk; sIdx[tid] = bi;
    __syncthreads();
    for (int o = 128; o > 0; o >>= 1) {
      if (tid < o && sKey[tid + o] > sKey[tid]) { sKey[tid] = sKey[tid + o]; sIdx[tid] = sIdx[tid + o]; }
      __syncthreads();
    }
    if (tid == 0) {
      const int id = (int)(~(unsigned int)sKey[0]);
      ids_out[q * KSEL + k] = (float)id;        // ids exact in fp32 (N < 2^24)
      selIds[q * KSEL + k]  = id;
      keys[sIdx[0]] = 0ull;                     // consume
    }
    __syncthreads();
  }
}

// -------------------------------- K5 ---------------------------------------
__global__ __launch_bounds__(64) void gather_kernel(
    const int* __restrict__ selIds, const float* __restrict__ item_emb,
    float* __restrict__ emb_out)
{
  const int b = blockIdx.x;                     // 256*100 rows
  const int id = selIds[b];
  const float4* src = (const float4*)(item_emb + (size_t)id * 256);
  float4* dst = (float4*)(emb_out + (size_t)b * 256);
  dst[threadIdx.x] = src[threadIdx.x];          // 64 lanes x float4 = 256 floats
}

// ------------------------------ launcher -----------------------------------
extern "C" void kernel_launch(void* const* d_in, const int* in_sizes, int n_in,
                              void* d_out, int out_size, void* d_ws, size_t ws_size,
                              hipStream_t stream)
{
  const float* hidden   = (const float*)d_in[0];
  const float* Wq       = (const float*)d_in[1];
  const float* bq       = (const float*)d_in[2];
  const float* gamma    = (const float*)d_in[3];
  const float* beta     = (const float*)d_in[4];
  const float* proj     = (const float*)d_in[5];
  const float* item_emb = (const float*)d_in[6];
  const int N   = in_sizes[6] / 256;
  const int NB2 = (N + 63) / 64;
  const int NC3 = (N + 2047) / 2048;
  const int nPad = NB2 * 64;

  float* qv_out  = (float*)d_out;                 // [256,256]
  float* ids_out = qv_out + 256 * 256;            // [256,100] (as float)
  float* emb_out = ids_out + 256 * KSEL;          // [256,100,256]

  size_t off = 0;
  auto carve = [&](size_t bytes) -> void* {
    void* p = (char*)d_ws + off;
    off += (bytes + 255) & ~(size_t)255;
    return p;
  };
  unsigned short* qv_bf     = (unsigned short*)carve((size_t)256 * 256 * 2);
  unsigned int*   qcodes    = (unsigned int*)carve(256 * 4);
  unsigned short* itemsBF   = (unsigned short*)carve((size_t)nPad * 256 * 2 + 32768); // +32KB slack for last-chunk async stage
  unsigned int*   itemCodes = (unsigned int*)carve((size_t)nPad * 4);
  float*          candScore = (float*)carve((size_t)256 * CAPC * 4);
  int*            candId    = (int*)carve((size_t)256 * CAPC * 4);
  int*            candCnt   = (int*)carve(256 * 4);
  int*            selIds    = (int*)carve((size_t)256 * KSEL * 4);

  hipMemsetAsync(candCnt, 0, 256 * sizeof(int), stream);

  qproj_kernel<<<16, 256, 0, stream>>>(hidden, Wq, bq, gamma, beta, proj,
                                       qv_out, qv_bf, qcodes);
  item_hash_kernel<<<NB2, 256, 0, stream>>>(item_emb, proj, itemsBF, itemCodes, N);
  dim3 g3(NC3, 16);
  score_kernel<<<g3, 256, 0, stream>>>(qv_bf, qcodes, itemsBF, itemCodes,
                                       candScore, candId, candCnt, N);
  select_kernel<<<256, 256, 0, stream>>>(candScore, candId, candCnt, ids_out, selIds);
  gather_kernel<<<256 * KSEL, 64, 0, stream>>>(selIds, item_emb, emb_out);
}
